// TransGCN_SP_10866267259410
// MI455X (gfx1250) — compile-verified, matrix operand order checked by reading, not run on hardware
//
#include <hip/hip_runtime.h>

#define NFEAT 128
#define LRELU 0.2f
#define WAVES_PER_BLOCK 4

typedef __bf16 v16bf __attribute__((ext_vector_type(16)));
typedef float  v8f   __attribute__((ext_vector_type(8)));

// K offset (within a 32-wide K chunk) for element e of a 16-element bf16
// A/B operand vector, per CDNA5 ISA 7.12.2 "16-bit A-Matrix 16x32":
//   lanes 0-15 (kh=0): V0..V3 hold K=0..7,  V4..V7 hold K=16..23
//   lanes 16-31 (kh=1): V0..V3 hold K=8..15, V4..V7 hold K=24..31
// Note: per lane this is two contiguous 8-element K runs: [kh*8, kh*8+8) and
// [16+kh*8, 16+kh*8+8) -> vectorizable as 4x float4.
__device__ __forceinline__ int kmap(int e, int kh) {
    return (e < 8) ? (kh * 8 + e) : (16 + kh * 8 + (e - 8));
}

__device__ __forceinline__ v8f wmma_bf16(v16bf a, v16bf b, v8f c) {
    return __builtin_amdgcn_wmma_f32_16x16x32_bf16(
        /*neg_a=*/false, a, /*neg_b=*/false, b,
        /*c_mod=*/(short)0, c, /*reuse_a=*/false, /*reuse_b=*/false);
}

// Pack 2x contiguous 8-float runs into a 16-wide bf16 operand vector.
__device__ __forceinline__ v16bf pack16(float4 f0, float4 f1, float4 f2, float4 f3) {
    v16bf a;
    a[0] = (__bf16)f0.x;  a[1] = (__bf16)f0.y;  a[2]  = (__bf16)f0.z;  a[3]  = (__bf16)f0.w;
    a[4] = (__bf16)f1.x;  a[5] = (__bf16)f1.y;  a[6]  = (__bf16)f1.z;  a[7]  = (__bf16)f1.w;
    a[8] = (__bf16)f2.x;  a[9] = (__bf16)f2.y;  a[10] = (__bf16)f2.z;  a[11] = (__bf16)f2.w;
    a[12] = (__bf16)f3.x; a[13] = (__bf16)f3.y; a[14] = (__bf16)f3.z;  a[15] = (__bf16)f3.w;
    return a;
}

// Load a 32-K chunk of a row-major fp32 row as a WMMA 16-bit A/B operand.
__device__ __forceinline__ v16bf load_row_chunk(const float* __restrict__ row, int kc, int kh) {
    const float* p = row + kc * 32;
    float4 f0 = *(const float4*)(p + kh * 8);
    float4 f1 = *(const float4*)(p + kh * 8 + 4);
    float4 f2 = *(const float4*)(p + 16 + kh * 8);
    float4 f3 = *(const float4*)(p + 16 + kh * 8 + 4);
    return pack16(f0, f1, f2, f3);
}

// ---------------------------------------------------------------------------
// One-shot weight packer: converts the 5 fp32 weight matrices into bf16 tiles
// already in per-lane WMMA B-operand order:
//   out[w][t = jt*4+kc][lane][e],  32 tiles x 32 lanes x 16 bf16 = 32KB/weight
// w=0..3: y = x @ W.T orientation (B[k][n] = W[j][k]); w=4: y = x @ W.
// ---------------------------------------------------------------------------
__global__ void pack_weights(const float* __restrict__ Wg1, const float* __restrict__ Wg2,
                             const float* __restrict__ Wb1, const float* __restrict__ Wb2,
                             const float* __restrict__ gcw, __bf16* __restrict__ out) {
    int tid = blockIdx.x * blockDim.x + threadIdx.x;   // one thread per (w, tile, lane)
    if (tid >= 5 * 32 * 32) return;
    int l  = tid & 31;
    int t  = (tid >> 5) & 31;
    int w  = tid >> 10;
    int jt = t >> 2, kc = t & 3;
    int n  = l & 15, kh = l >> 4;
    int j  = jt * 16 + n;
    const float* W = (w == 0) ? Wg1 : (w == 1) ? Wg2 : (w == 2) ? Wb1 : (w == 3) ? Wb2 : gcw;
    v16bf b;
#pragma unroll
    for (int e = 0; e < 16; ++e) {
        int k = kc * 32 + kmap(e, kh);
        b[e] = (w < 4) ? (__bf16)W[j * NFEAT + k]      // B[k][n] = W[j][k]  (x @ W.T)
                       : (__bf16)W[k * NFEAT + j];     // B[k][n] = W[k][j]  (x @ W)
    }
    ((v16bf*)out)[tid] = b;
}

// Fetch a pre-packed B tile: one contiguous 32B vector load per lane.
__device__ __forceinline__ v16bf load_packed_b(const __bf16* __restrict__ pk, int jt, int kc, int lane) {
    return ((const v16bf*)pk)[(jt * 4 + kc) * 32 + lane];
}

__global__ void zero_f32(float* __restrict__ p, long n) {
    long i = blockIdx.x * (long)blockDim.x + threadIdx.x;
    if (i < n) p[i] = 0.0f;
}

// Scatter SpMM: dst[row] += val * src[col], one thread per (edge, 4 features)
__global__ void spmm_scatter(const int* __restrict__ rows, const int* __restrict__ cols,
                             const float* __restrict__ vals, const float* __restrict__ src,
                             float* __restrict__ dst, int nedge) {
    long tid = blockIdx.x * (long)blockDim.x + threadIdx.x;
    int e = (int)(tid >> 5);
    if (e >= nedge) return;
    int c4 = ((int)tid & 31) * 4;
    float v = vals[e];
    int r = rows[e], c = cols[e];
    float4 xv = *(const float4*)&src[(size_t)c * NFEAT + c4];
    float* d = &dst[(size_t)r * NFEAT + c4];
    unsafeAtomicAdd(d + 0, v * xv.x);
    unsafeAtomicAdd(d + 1, v * xv.y);
    unsafeAtomicAdd(d + 2, v * xv.z);
    unsafeAtomicAdd(d + 3, v * xv.w);
}

// Fused FiLM + 5 GEMMs via bf16 WMMA. One wave handles a 16-node tile.
__global__ __launch_bounds__(WAVES_PER_BLOCK * 32)
void fused_film_gemm(const float* __restrict__ x, const float* __restrict__ nb,
                     const __bf16* __restrict__ wpk,   // packed weights, 32KB each
                     const float* __restrict__ r,
                     float* __restrict__ out_hk, float* __restrict__ out_m,
                     float* __restrict__ xw_ws, int numTiles) {
    __shared__ float lds_m[WAVES_PER_BLOCK * 16 * NFEAT];

    const int lane = threadIdx.x & 31;
    const int wave = threadIdx.x >> 5;
    const int tile = blockIdx.x * WAVES_PER_BLOCK + wave;
    const bool active = tile < numTiles;     // wave-uniform
    const int row0 = tile * 16;
    const int mrow = lane & 15;              // A-layout M / D-layout N index
    const int kh   = lane >> 4;              // lane half
    float* lm = &lds_m[wave * 16 * NFEAT];

    const __bf16* Bg1 = wpk;
    const __bf16* Bg2 = wpk + 1 * 32 * 32 * 16;
    const __bf16* Bb1 = wpk + 2 * 32 * 32 * 16;
    const __bf16* Bb2 = wpk + 3 * 32 * 32 * 16;
    const __bf16* Bgc = wpk + 4 * 32 * 32 * 16;

    v16bf xA[4], nA[4];
    if (active) {
        const float* xrow = x  + (size_t)(row0 + mrow) * NFEAT;
        const float* nrow = nb + (size_t)(row0 + mrow) * NFEAT;
        for (int kc = 0; kc < 4; ++kc) {
            xA[kc] = load_row_chunk(xrow, kc, kh);
            nA[kc] = load_row_chunk(nrow, kc, kh);
        }

        // Pass 1: gamma/beta GEMMs + FiLM epilogue -> m (to d_out and LDS)
        for (int jt = 0; jt < 8; ++jt) {
            v8f gacc = {}, bacc = {};
            for (int kc = 0; kc < 4; ++kc) {
                gacc = wmma_bf16(xA[kc], load_packed_b(Bg1, jt, kc, lane), gacc);
                gacc = wmma_bf16(nA[kc], load_packed_b(Bg2, jt, kc, lane), gacc);
                bacc = wmma_bf16(xA[kc], load_packed_b(Bb1, jt, kc, lane), bacc);
                bacc = wmma_bf16(nA[kc], load_packed_b(Bb2, jt, kc, lane), bacc);
            }
            const int col = jt * 16 + mrow;  // D-layout: N = lane%16
            const float rv = r[col];
#pragma unroll
            for (int v = 0; v < 8; ++v) {    // D-layout: M = v + 8*(lane/16)
                int m = v + 8 * kh;
                float g  = (gacc[v] > 0.0f ? gacc[v] : LRELU * gacc[v]) + 1.0f;
                float bt = (bacc[v] > 0.0f ? bacc[v] : LRELU * bacc[v]);
                size_t gi = (size_t)(row0 + m) * NFEAT + col;
                float mval = x[gi] + (g * rv + bt) - nb[gi];
                out_m[gi] = mval;
                lm[m * NFEAT + col] = mval;  // LDS transpose staging
            }
        }
    }
    __syncthreads();  // m tile visible for A-layout readback

    if (active) {
        // Pass 2: h_s = m @ gc_w and xw = x @ gc_w; seed h_k = xw + h_s
        v16bf mA[4];
        for (int kc = 0; kc < 4; ++kc)
            mA[kc] = load_row_chunk(&lm[mrow * NFEAT], kc, kh);   // ds_load_b128 x4

        for (int jt = 0; jt < 8; ++jt) {
            v8f hacc = {}, wacc = {};
            for (int kc = 0; kc < 4; ++kc) {
                v16bf b = load_packed_b(Bgc, jt, kc, lane);
                hacc = wmma_bf16(mA[kc], b, hacc);
                wacc = wmma_bf16(xA[kc], b, wacc);
            }
            const int col = jt * 16 + mrow;
#pragma unroll
            for (int v = 0; v < 8; ++v) {
                int m = v + 8 * kh;
                size_t gi = (size_t)(row0 + m) * NFEAT + col;
                xw_ws[gi]  = wacc[v];
                out_hk[gi] = wacc[v] + hacc[v];  // h_k seed: xw + h_s
            }
        }
    }
}

__global__ void finalize_hk(float* __restrict__ hk, const float* __restrict__ norm, long n) {
    long i = blockIdx.x * (long)blockDim.x + threadIdx.x;
    if (i < n) hk[i] = hk[i] / (norm[i >> 7] + 1.0f);
}

extern "C" void kernel_launch(void* const* d_in, const int* in_sizes, int n_in,
                              void* d_out, int out_size, void* d_ws, size_t ws_size,
                              hipStream_t stream) {
    const float* x    = (const float*)d_in[0];
    const int*   erow = (const int*)d_in[1];
    const int*   ecol = (const int*)d_in[2];
    const float* eval_= (const float*)d_in[3];
    const float* norm = (const float*)d_in[4];
    const float* Wg1  = (const float*)d_in[5];
    const float* Wg2  = (const float*)d_in[6];
    const float* Wb1  = (const float*)d_in[7];
    const float* Wb2  = (const float*)d_in[8];
    const float* r    = (const float*)d_in[9];
    const float* gcw  = (const float*)d_in[10];

    const int n     = in_sizes[0] / NFEAT;   // 100000 nodes
    const int nedge = in_sizes[1];           // 1.6M edges
    const long nelem = (long)n * NFEAT;

    float*  nb     = (float*)d_ws;           // neighbor [N,128]
    float*  xw     = nb + nelem;             // xw       [N,128]
    __bf16* wpk    = (__bf16*)(xw + nelem);  // packed bf16 weights, 5 x 32KB
    float*  out_hk = (float*)d_out;          // h_k      [N,128]
    float*  out_m  = out_hk + nelem;         // m        [N,128]

    const int ZB = 256;

    // One-shot: pack weights into WMMA B-operand order (bf16).
    pack_weights<<<(5 * 32 * 32 + ZB - 1) / ZB, ZB, 0, stream>>>(Wg1, Wg2, Wb1, Wb2, gcw, wpk);

    zero_f32<<<(int)((nelem + ZB - 1) / ZB), ZB, 0, stream>>>(nb, nelem);

    // neighbor = spmm(adj, x)
    long sthreads = (long)nedge * 32;
    spmm_scatter<<<(int)((sthreads + 255) / 256), 256, 0, stream>>>(erow, ecol, eval_, x, nb, nedge);

    // fused FiLM + GEMMs (bf16 WMMA): m, xw, h_k seed = xw + h_s
    int numTiles = (n + 15) / 16;            // N % 16 == 0 here
    fused_film_gemm<<<(numTiles + WAVES_PER_BLOCK - 1) / WAVES_PER_BLOCK,
                      WAVES_PER_BLOCK * 32, 0, stream>>>(
        x, nb, wpk, r, out_hk, out_m, xw, numTiles);

    // h_k += spmm(adj, xw)
    spmm_scatter<<<(int)((sthreads + 255) / 256), 256, 0, stream>>>(erow, ecol, eval_, xw, out_hk, nedge);

    // h_k /= (norm + 1)
    finalize_hk<<<(int)((nelem + ZB - 1) / ZB), ZB, 0, stream>>>(out_hk, norm, nelem);
}
